// CausalSelfAttention_8280696946882
// MI455X (gfx1250) — matrix operand order repaired and checked by reading, not compiled, not run on hardware
//
#include <hip/hip_runtime.h>
#include <hip/hip_bf16.h>
#include <stdint.h>

typedef _Float16 f16;
typedef __attribute__((ext_vector_type(8)))  _Float16 v8h;
typedef __attribute__((ext_vector_type(16))) _Float16 v16h;
typedef __attribute__((ext_vector_type(8)))  float    v8f;

#define CAT16(lo, hi) __builtin_shufflevector(lo, hi, 0,1,2,3,4,5,6,7,8,9,10,11,12,13,14,15)

#define B_   8
#define T_   2048
#define C_   126
#define H_   6
#define HD_  21
#define HDP_ 32
#define N3_  378
#define N3P_ 384
#define CP_  128
#define BT_  (B_*T_)

// workspace layout (bytes)
#define OFF_WAT 0
#define SZ_WAT  (N3P_*CP_*2)          // 98304  : w_attn^T f16 padded [384][128]
#define OFF_WPT (OFF_WAT + SZ_WAT)
#define SZ_WPT  (CP_*CP_*2)           // 32768  : w_proj^T f16 padded [128][128]
#define OFF_QP  (OFF_WPT + SZ_WPT)
#define SZ_QP   (B_*H_*T_*HDP_*2)     // 6291456: Q [B,H,T,32] f16
#define OFF_KP  (OFF_QP + SZ_QP)      //          K [B,H,T,32] f16
#define OFF_VT  (OFF_KP + SZ_QP)      //          V^T [B,H,32,T] f16
#define OFF_AO  (OFF_VT + SZ_QP)
#define SZ_AO   (BT_*CP_*2)           // 4194304: attn out [B*T,128] f16
#define ZERO_BYTES (3*SZ_QP + SZ_AO)  // pad lanes must be zero for WMMA K-dim

static __device__ inline v8f wmma_f16(v16h a, v16h b, v8f c) {
  // v_wmma_f32_16x16x32_f16: (neg_a, A, neg_b, B, c_mod, C, reuse_a, reuse_b)
  return __builtin_amdgcn_wmma_f32_16x16x32_f16(false, a, false, b, (short)0, c, false, false);
}

// A-fragment (16-bit, 16x32): lane half 0 holds K {0..15}, half 1 holds
// K {16..31} -> two contiguous 16B chunks at +half*16 and +half*16+8.
static __device__ inline v16h load_afrag(const f16* rowbase, int half) {
  v8h lo = *(const v8h*)(rowbase + half*16);
  v8h hi = *(const v8h*)(rowbase + half*16 + 8);
  return CAT16(lo, hi);
}

// ---------------------------------------------------------------- prep
__global__ void zero_ws_kernel(uint32_t* __restrict__ p, int n) {
  int i = blockIdx.x*blockDim.x + threadIdx.x;
  int stride = gridDim.x*blockDim.x;
  for (; i < n; i += stride) p[i] = 0u;
}

__global__ void prep_w_kernel(const float* __restrict__ wa, const float* __restrict__ wp,
                              f16* __restrict__ waT, f16* __restrict__ wpT) {
  int i = blockIdx.x*blockDim.x + threadIdx.x;
  int stride = gridDim.x*blockDim.x;
  const int tot = N3P_*CP_ + CP_*CP_;
  for (; i < tot; i += stride) {
    if (i < N3P_*CP_) {
      int n = i / CP_, k = i % CP_;
      waT[i] = (n < N3_ && k < C_) ? (f16)wa[k*N3_ + n] : (f16)0.f;
    } else {
      int j = i - N3P_*CP_;
      int n = j / CP_, k = j % CP_;
      wpT[j] = (n < C_ && k < C_) ? (f16)wp[k*C_ + n] : (f16)0.f;
    }
  }
}

// ---------------------------------------------------------------- QKV projection
// grid: 1024 blocks x 128 threads (4 waves). Each block: one 16-row tile of x,
// each wave: 6 of the 24 N-tiles (N padded 378->384). K = 126 padded to 128.
__global__ void qkv_kernel(const float* __restrict__ x, const f16* __restrict__ waT,
                           f16* __restrict__ Qp, f16* __restrict__ Kp, f16* __restrict__ VT) {
  __shared__ __align__(32) f16 xs[16*CP_];
  const int m0  = blockIdx.x * 16;
  const int tid = threadIdx.x;
  for (int i = tid; i < 16*CP_; i += 128) {
    int row = i >> 7, k = i & 127;
    float v = (k < C_) ? x[(size_t)(m0+row)*C_ + k] : 0.f;
    xs[i] = (f16)v;
  }
  __syncthreads();
  const int wave = tid >> 5, lane = tid & 31, half = lane >> 4, nl = lane & 15;

  v16h af[4];                               // A fragments from LDS (ds_load_b128 x2)
  #pragma unroll
  for (int kc = 0; kc < 4; ++kc)
    af[kc] = load_afrag(xs + nl*CP_ + kc*32, half);

  for (int t = wave*6; t < wave*6 + 6; ++t) {
    const f16* wrow = waT + (size_t)(t*16+nl)*CP_ + half*16;
    v16h bf0 = *(const v16h*)(wrow);        // preload all 4 K-chunk B fragments
    v16h bf1 = *(const v16h*)(wrow + 32);   // so WMMAs pipeline behind the loads
    v16h bf2 = *(const v16h*)(wrow + 64);
    v16h bf3 = *(const v16h*)(wrow + 96);
    v8f acc = {};
    acc = wmma_f16(af[0], bf0, acc);
    acc = wmma_f16(af[1], bf1, acc);
    acc = wmma_f16(af[2], bf2, acc);
    acc = wmma_f16(af[3], bf3, acc);

    int c = t*16 + nl;                      // qkv column (uniform over r)
    if (c < N3_) {
      int sec = c / C_;                     // 0=Q 1=K 2=V
      int cc  = c % C_;
      int h = cc / HD_, d = cc % HD_;
      #pragma unroll
      for (int r = 0; r < 8; ++r) {
        int g = m0 + r + 8*half;            // global row in [B*T]
        int b = g / T_, tt = g % T_;
        int bh = b*H_ + h;
        f16 val = (f16)acc[r];
        if (sec == 0)      Qp[((size_t)bh*T_ + tt)*HDP_ + d] = val;
        else if (sec == 1) Kp[((size_t)bh*T_ + tt)*HDP_ + d] = val;
        else               VT[((size_t)bh*HDP_ + d)*T_ + tt] = val;
      }
    }
  }
}

// ---------------------------------------------------------------- flash attention
// grid: B*H*(T/64) blocks x 128 threads. Each wave owns one 16-query tile and
// streams 32-key chunks with online (running max/sum) causal softmax.
__global__ void attn_kernel(const f16* __restrict__ Qp, const f16* __restrict__ Kp,
                            const f16* __restrict__ VT, f16* __restrict__ AO) {
  __shared__ __align__(32) f16 pls[4][16*32]; // per-wave P staging (D->A layout)
  const int bh   = blockIdx.x >> 5;
  const int qblk = blockIdx.x & 31;
  const int b = bh / H_, h = bh % H_;
  const int tid = threadIdx.x;
  const int wave = tid >> 5, lane = tid & 31, half = lane >> 4, nl = lane & 15;
  const int q0 = qblk*64 + wave*16;

  const f16* Qb = Qp + (size_t)bh*T_*HDP_;
  const f16* Kb = Kp + (size_t)bh*T_*HDP_;
  const f16* Vb = VT + (size_t)bh*HDP_*T_;

  const v16h qa = load_afrag(Qb + (size_t)(q0+nl)*HDP_, half);

  float mrow[8], lrow[8];
  #pragma unroll
  for (int r = 0; r < 8; ++r) { mrow[r] = -1e30f; lrow[r] = 0.f; }
  v8f o0 = {}, o1 = {};
  const float scale = 0.2182178902f;        // 1/sqrt(21)

  const int nch = (q0 + 15)/32 + 1;         // causal: keys [0, q0+16)
  for (int c = 0; c < nch; ++c) {
    const int kb0 = c*32;
    if (c + 1 < nch) {                      // prefetch next K and V chunks
      __builtin_prefetch(Kb + (size_t)(kb0 + 32)*HDP_, 0, 1);
      __builtin_prefetch(Vb + (size_t)nl*T_ + kb0 + 32, 0, 1);
    }
    // B fragments: K^T tiles (key-major rows are contiguous 32B per lane)
    v16h kf0 = *(const v16h*)(Kb + (size_t)(kb0 + nl)*HDP_      + half*16);
    v16h kf1 = *(const v16h*)(Kb + (size_t)(kb0 + 16 + nl)*HDP_ + half*16);
    // V B fragments issued early: latency hides behind S-WMMA + softmax VALU
    v16h vf0 = *(const v16h*)(Vb + (size_t)nl*T_        + kb0 + half*16);
    v16h vf1 = *(const v16h*)(Vb + (size_t)(16+nl)*T_   + kb0 + half*16);

    v8f z = {};
    v8f s0 = wmma_f16(qa, kf0, z);
    v8f s1 = wmma_f16(qa, kf1, z);

    #pragma unroll
    for (int r = 0; r < 8; ++r) {           // causal mask + online softmax update
      int qg = q0 + r + 8*half;
      float s0v = (kb0 + nl      <= qg) ? s0[r]*scale : -1e30f;
      float s1v = (kb0 + 16 + nl <= qg) ? s1[r]*scale : -1e30f;
      float mx = fmaxf(s0v, s1v);
      #pragma unroll
      for (int off = 1; off < 16; off <<= 1)
        mx = fmaxf(mx, __shfl_xor(mx, off, 32));
      float mnew = fmaxf(mrow[r], mx);
      float corr = __expf(mrow[r] - mnew);
      mrow[r] = mnew;
      float p0 = __expf(s0v - mnew);
      float p1 = __expf(s1v - mnew);
      float ps = p0 + p1;
      #pragma unroll
      for (int off = 1; off < 16; off <<= 1)
        ps += __shfl_xor(ps, off, 32);
      lrow[r] = lrow[r]*corr + ps;
      o0[r] *= corr;
      o1[r] *= corr;
      int m = r + 8*half;
      pls[wave][m*32 + nl]      = (f16)p0;  // stage P for A-layout reload
      pls[wave][m*32 + 16 + nl] = (f16)p1;
    }
    __builtin_amdgcn_wave_barrier();        // LDS in-order per wave; stop reordering
    v16h pa = load_afrag(pls[wave] + nl*32, half);
    __builtin_amdgcn_wave_barrier();
    o0 = wmma_f16(pa, vf0, o0);             // O += P @ V
    o1 = wmma_f16(pa, vf1, o1);
  }
  #pragma unroll
  for (int r = 0; r < 8; ++r) {             // normalize + write head slice
    float inv = 1.0f / lrow[r];
    int q = q0 + r + 8*half;
    size_t rowbase = ((size_t)b*T_ + q)*CP_ + h*HD_;
    AO[rowbase + nl] = (f16)(o0[r]*inv);    // d = 0..15 (< 21)
    if (nl < 5) AO[rowbase + 16 + nl] = (f16)(o1[r]*inv);  // d = 16..20
  }
}

// ---------------------------------------------------------------- output projection
__global__ void proj_kernel(const f16* __restrict__ AO, const f16* __restrict__ wpT,
                            float* __restrict__ out) {
  const int m0  = blockIdx.x * 16;
  const int tid = threadIdx.x;
  const int wave = tid >> 5, lane = tid & 31, half = lane >> 4, nl = lane & 15;
  v16h af[4];
  #pragma unroll
  for (int kc = 0; kc < 4; ++kc)
    af[kc] = load_afrag(AO + (size_t)(m0+nl)*CP_ + kc*32, half);

  for (int t = wave*2; t < wave*2 + 2; ++t) {  // 8 N-tiles (126 padded to 128)
    const f16* wrow = wpT + (size_t)(t*16+nl)*CP_ + half*16;
    v16h bf0 = *(const v16h*)(wrow);
    v16h bf1 = *(const v16h*)(wrow + 32);
    v16h bf2 = *(const v16h*)(wrow + 64);
    v16h bf3 = *(const v16h*)(wrow + 96);
    v8f acc = {};
    acc = wmma_f16(af[0], bf0, acc);
    acc = wmma_f16(af[1], bf1, acc);
    acc = wmma_f16(af[2], bf2, acc);
    acc = wmma_f16(af[3], bf3, acc);
    #pragma unroll
    for (int r = 0; r < 8; ++r) {
      int g  = m0 + r + 8*half;
      int cc = t*16 + nl;
      if (cc < C_) out[(size_t)g*C_ + cc] = acc[r];
    }
  }
}

// ---------------------------------------------------------------- launch
extern "C" void kernel_launch(void* const* d_in, const int* in_sizes, int n_in,
                              void* d_out, int out_size, void* d_ws, size_t ws_size,
                              hipStream_t stream) {
  const float* x  = (const float*)d_in[0];
  const float* wa = (const float*)d_in[1];
  const float* wp = (const float*)d_in[2];
  char* ws = (char*)d_ws;
  f16* waT = (f16*)(ws + OFF_WAT);
  f16* wpT = (f16*)(ws + OFF_WPT);
  f16* Qp  = (f16*)(ws + OFF_QP);
  f16* Kp  = (f16*)(ws + OFF_KP);
  f16* VT  = (f16*)(ws + OFF_VT);
  f16* AO  = (f16*)(ws + OFF_AO);

  zero_ws_kernel<<<2048, 256, 0, stream>>>((uint32_t*)(ws + OFF_QP), ZERO_BYTES/4);
  prep_w_kernel<<<256, 256, 0, stream>>>(wa, wp, waT, wpT);
  qkv_kernel<<<BT_/16, 128, 0, stream>>>(x, waT, Qp, Kp, VT);
  attn_kernel<<<B_*H_*(T_/64), 128, 0, stream>>>(Qp, Kp, VT, AO);
  proj_kernel<<<BT_/16, 128, 0, stream>>>(AO, wpT, (float*)d_out);
}
